// Hete_GNNs_87814901334299
// MI455X (gfx1250) — matrix operand order repaired and checked by reading, not compiled, not run on hardware
//
#include <hip/hip_runtime.h>
#include <math.h>

#define EMB 300
#define EMBP 320           // EMB padded to multiple of 32 for guard-free WMMA K-loop
#define HID 512
#define G3 1536
#define BB 32
#define NN 80
#define LAQ 8
#define LLQ 40
#define KE 6
#define GH 16
#define AH 32
#define DHH 16
#define POL 3
#define GATF (GH*HID)      // 8192
#define NNODE (BB*NN)      // 2560
#define NE (NNODE*KE)      // 15360

typedef __bf16 bf16_t;
typedef bf16_t  v8bf  __attribute__((ext_vector_type(8)));
typedef bf16_t  v16bf __attribute__((ext_vector_type(16)));
typedef float   v8f   __attribute__((ext_vector_type(8)));

// pack two f32 into one dword of 2x bf16 (truncate) with a single v_perm_b32
__device__ __forceinline__ unsigned pkbf(float lo, float hi) {
  return __builtin_amdgcn_perm(__builtin_bit_cast(unsigned, hi),
                               __builtin_bit_cast(unsigned, lo), 0x07060302u);
}
__device__ __forceinline__ bf16_t f2bf(float x) {
  unsigned u = __builtin_bit_cast(unsigned, x);
  return __builtin_bit_cast(bf16_t, (unsigned short)(u >> 16));
}
__device__ __forceinline__ float sigm(float x) { return 1.0f / (1.0f + expf(-x)); }

// ---------------------------------------------------------------------------
// WMMA GEMM:  C[M,N] = A[M,K] @ W[N,K]^T + bias[N]
// A, W pre-converted bf16. Requirements: M%16==0, N%64==0, K%32==0.
// One wave per 16x64 output strip: A-frag (2x b128) loaded once per K-step,
// 4 B-frags (2x b128 each), 4 WMMAs. No conversions, no divergent guards.
// ---------------------------------------------------------------------------
__global__ void k_wmma_gemm(const bf16_t* __restrict__ A, int lda,
                            const bf16_t* __restrict__ W, int ldw,
                            const float* __restrict__ bias,
                            float* __restrict__ C, int ldc,
                            int N, int K) {
  const int lane = threadIdx.x;      // 0..31, full wave32 -> EXEC all ones
  const int g    = lane >> 4;        // half-wave group
  const int ml   = lane & 15;
  const int m0   = blockIdx.y * 16;
  const int n0   = blockIdx.x * 64;
  const bf16_t* Arow = A + (size_t)(m0 + ml) * lda;
  const bf16_t* Wrow = W + (size_t)(n0 + ml) * ldw;
  v8f acc[4] = {{}, {}, {}, {}};
  for (int k0 = 0; k0 < K; k0 += 32) {
    // A-frag: lane holds row m0+ml, K in [g*16, g*16+16)
    v8bf alo = *(const v8bf*)(Arow + k0 + g * 16);
    v8bf ahi = *(const v8bf*)(Arow + k0 + g * 16 + 8);
    v16bf av = __builtin_shufflevector(alo, ahi, 0, 1, 2, 3, 4, 5, 6, 7,
                                       8, 9, 10, 11, 12, 13, 14, 15);
#pragma unroll
    for (int i = 0; i < 4; ++i) {
      // B-frag i: lane holds column n0+i*16+ml, K in [g*16, g*16+16)
      v16bf bv = *(const v16bf*)(Wrow + (size_t)i * 16 * ldw + k0 + g * 16);
      acc[i] = __builtin_amdgcn_wmma_f32_16x16x32_bf16(false, av, false, bv,
                                                       (short)0, acc[i], false, false);
    }
  }
#pragma unroll
  for (int i = 0; i < 4; ++i) {
    int n = n0 + i * 16 + ml;
    float bv = bias ? bias[n] : 0.0f;
#pragma unroll
    for (int r = 0; r < 8; ++r) {
      int mm = m0 + r + 8 * g;       // C/D layout: VGPR r -> row r + 8*g
      C[(size_t)mm * ldc + n] = acc[i][r] + bv;
    }
  }
}

static void gemm(const bf16_t* A, int lda, const bf16_t* W, int ldw,
                 const float* bias, float* C, int ldc, int M, int N, int K,
                 hipStream_t s) {
  dim3 grid(N / 64, M / 16);
  k_wmma_gemm<<<grid, 32, 0, s>>>(A, lda, W, ldw, bias, C, ldc, N, K);
}

// f32 -> bf16 bulk convert, two elements per thread (one v_perm + b32 store)
__global__ void k_cvt(const float* __restrict__ src, bf16_t* __restrict__ dst,
                      int npair) {
  int i = blockIdx.x * blockDim.x + threadIdx.x;
  if (i >= npair) return;
  ((unsigned*)dst)[i] = pkbf(src[2 * i], src[2 * i + 1]);
}

static void cvt(const float* src, bf16_t* dst, int n, hipStream_t s) {
  int npair = n / 2;
  k_cvt<<<(npair + 255) / 256, 256, 0, s>>>(src, dst, npair);
}

// ---------------------------------------------------------------------------
__global__ void k_lengths(const int* __restrict__ text, const int* __restrict__ aspect,
                          const int* __restrict__ left,
                          int* tlen, int* alen, int* llen) {
  int b = threadIdx.x;
  if (b >= BB) return;
  int c = 0;
  for (int t = 0; t < NN; ++t) c += (text[b * NN + t] != 0);
  tlen[b] = c + 5;
  c = 0;
  for (int t = 0; t < LAQ; ++t) c += (aspect[b * LAQ + t] != 0);
  alen[b] = c;
  c = 0;
  for (int t = 0; t < LLQ; ++t) c += (left[b * LLQ + t] != 0);
  llen[b] = c;
}

// embedding gather into K-padded bf16 rows (zero tail)
__global__ void k_embed(const float* __restrict__ emb, const int* __restrict__ idx,
                        bf16_t* __restrict__ dst) {
  int r = blockIdx.x;
  int tok = idx[r];
  const float* src = emb + (size_t)tok * EMB;
  unsigned* drow = (unsigned*)dst + (size_t)r * (EMBP / 2);
  for (int p = threadIdx.x; p < EMBP / 2; p += blockDim.x)
    drow[p] = (p < EMB / 2) ? pkbf(src[2 * p], src[2 * p + 1]) : 0u;
}

// pad+convert Wih [G3,EMB] -> bf16 [G3,EMBP] with zero tail
__global__ void k_pad_w(const float* __restrict__ src, bf16_t* __restrict__ dst) {
  int r = blockIdx.x;
  const float* srow = src + (size_t)r * EMB;
  unsigned* drow = (unsigned*)dst + (size_t)r * (EMBP / 2);
  for (int p = threadIdx.x; p < EMBP / 2; p += blockDim.x)
    drow[p] = (p < EMB / 2) ? pkbf(srow[2 * p], srow[2 * p + 1]) : 0u;
}

__global__ void k_gru_step(const float* __restrict__ xg, const float* __restrict__ hg,
                           float* __restrict__ h, bf16_t* __restrict__ h_bf,
                           float* __restrict__ out, bf16_t* __restrict__ out_bf,
                           const int* __restrict__ lens, int t, int T) {
  int b = blockIdx.x;
  bool valid = (t < lens[b]);
  const float* xgr = xg + ((size_t)b * T + t) * G3;
  const float* hgr = hg + (size_t)b * G3;
  for (int c = threadIdx.x; c < HID; c += blockDim.x) {
    float r  = sigm(xgr[c] + hgr[c]);
    float z  = sigm(xgr[HID + c] + hgr[HID + c]);
    float nn = tanhf(xgr[2 * HID + c] + r * hgr[2 * HID + c]);
    float hp = h[(size_t)b * HID + c];
    float hn = (1.0f - z) * nn + z * hp;
    float hout = valid ? hn : hp;
    float oout = valid ? hn : 0.0f;
    h[(size_t)b * HID + c] = hout;
    h_bf[(size_t)b * HID + c] = f2bf(hout);
    out[((size_t)b * T + t) * HID + c] = oout;
    out_bf[((size_t)b * T + t) * HID + c] = f2bf(oout);
  }
}

__global__ void k_pool_mean(const float* __restrict__ src, const int* __restrict__ lens,
                            bf16_t* __restrict__ dst, int T) {
  int b = blockIdx.x;
  float inv = 1.0f / (float)lens[b];
  for (int hh = threadIdx.x; hh < HID; hh += blockDim.x) {
    float s = 0.0f;
    for (int t = 0; t < T; ++t) s += src[((size_t)b * T + t) * HID + hh];
    dst[(size_t)b * HID + hh] = f2bf(s * inv);
  }
}

__global__ void k_qw(const float* __restrict__ qx, const float* __restrict__ Wbl,
                     float* __restrict__ qw) {
  int i = blockIdx.x * blockDim.x + threadIdx.x;
  if (i >= BB * HID) return;
  int b = i / HID, he = i % HID;
  int hh = he / DHH, e = he % DHH;
  float s = 0.0f;
  for (int d = 0; d < DHH; ++d)
    s += qx[(size_t)b * HID + hh * DHH + d] * Wbl[d * DHH + e];
  qw[i] = s;
}

__global__ void k_att_score(const float* __restrict__ qw, const float* __restrict__ kx,
                            bf16_t* __restrict__ attout, int Lk) {
  __shared__ float sl[NN];
  __shared__ float sa[NN];
  int bh = blockIdx.x;
  int b = bh >> 5, hh = bh & 31;
  int tid = threadIdx.x;
  if (tid < Lk) {
    const float* kr = kx + ((size_t)b * Lk + tid) * HID + hh * DHH;
    const float* qr = qw + (size_t)b * HID + hh * DHH;
    float s = 0.0f;
    for (int d = 0; d < DHH; ++d) s += qr[d] * kr[d];
    sl[tid] = s;
  }
  __syncthreads();
  if (tid == 0) {
    float mx = -1e30f;
    for (int k = 0; k < Lk; ++k) mx = fmaxf(mx, sl[k]);
    float den = 0.0f;
    for (int k = 0; k < Lk; ++k) { float e = expf(sl[k] - mx); sa[k] = e; den += e; }
    float inv = 1.0f / den;
    for (int k = 0; k < Lk; ++k) sa[k] *= inv;
  }
  __syncthreads();
  if (tid < DHH) {
    float o = 0.0f;
    for (int k = 0; k < Lk; ++k)
      o += sa[k] * kx[((size_t)b * Lk + k) * HID + hh * DHH + tid];
    attout[(size_t)b * HID + hh * DHH + tid] = f2bf(o);
  }
}

__global__ void k_edge_e(const float* __restrict__ xl, const float* __restrict__ xr,
                         const int* __restrict__ esrc, const int* __restrict__ edst,
                         const float* __restrict__ attw, float* __restrict__ ebuf) {
  __shared__ float red[256];
  int e = blockIdx.x;
  int s = esrc[e], d = edst[e];
  int tid = threadIdx.x;
  for (int hh = 0; hh < GH; ++hh) {
    float part = 0.0f;
    const float* xls = xl + (size_t)s * GATF + hh * HID;
    const float* xrd = xr + (size_t)d * GATF + hh * HID;
    const float* aw  = attw + hh * HID;
    for (int c = tid; c < HID; c += 256) {
      float v = xls[c] + xrd[c];
      v = (v >= 0.0f) ? v : 0.2f * v;
      part += v * aw[c];
    }
    red[tid] = part;
    __syncthreads();
    for (int off = 128; off > 0; off >>= 1) {
      if (tid < off) red[tid] += red[tid + off];
      __syncthreads();
    }
    if (tid == 0) ebuf[(size_t)e * GH + hh] = red[0];
    __syncthreads();
  }
}

// Edges for dst node d are exactly indices d*KE .. d*KE+KE-1 (from setup layout),
// so segment max/sum are per-node local: no atomics needed.
__global__ void k_gat_agg(const float* __restrict__ xl, const float* __restrict__ ebuf,
                          const int* __restrict__ esrc, const float* __restrict__ gbias,
                          float* __restrict__ gx) {
  __shared__ float sal[GH * KE];
  __shared__ int ssrc[KE];
  int node = blockIdx.x;
  int tid = threadIdx.x;
  if (tid < KE) ssrc[tid] = esrc[node * KE + tid];
  if (tid < GH) {
    int hh = tid;
    float mx = -1e30f;
    for (int k = 0; k < KE; ++k)
      mx = fmaxf(mx, ebuf[(size_t)(node * KE + k) * GH + hh]);
    float ex[KE];
    float den = 0.0f;
    for (int k = 0; k < KE; ++k) {
      ex[k] = expf(ebuf[(size_t)(node * KE + k) * GH + hh] - mx);
      den += ex[k];
    }
    float inv = 1.0f / den;
    for (int k = 0; k < KE; ++k) sal[hh * KE + k] = ex[k] * inv;
  }
  __syncthreads();
  for (int c = tid; c < HID; c += blockDim.x) {
    float acc = 0.0f;
    for (int hh = 0; hh < GH; ++hh)
      for (int k = 0; k < KE; ++k)
        acc += xl[(size_t)ssrc[k] * GATF + hh * HID + c] * sal[hh * KE + k];
    float v = acc * (1.0f / GH) + gbias[c];
    gx[(size_t)node * HID + c] = (v >= 0.0f) ? v : 0.01f * v;
  }
}

__global__ void k_cmask(const float* __restrict__ ctx, const float* __restrict__ adj,
                        const int* __restrict__ llen, const int* __restrict__ alen,
                        float* __restrict__ cm) {
  int idx = blockIdx.x;            // b*NN + n
  int b = idx / NN, n = idx % NN;
  float asum = 0.0f;
  for (int j = NN - 5; j < NN; ++j) asum += adj[(size_t)idx * NN + j];
  int st = llen[b], en = llen[b] + alen[b] - 1;
  bool span  = (n >= st) && (n <= en);
  bool extra = (n < NN - 5) && (asum != 0.0f);
  float msk = (span || extra) ? 1.0f : 0.0f;
  for (int hh = threadIdx.x; hh < HID; hh += blockDim.x)
    cm[(size_t)idx * HID + hh] = msk * ctx[(size_t)idx * HID + hh];
}

__global__ void k_xsum(const float* __restrict__ gx, const int* __restrict__ llen,
                       const int* __restrict__ alen, float* __restrict__ xsum) {
  int b = blockIdx.x;
  int st = llen[b], en = llen[b] + alen[b] - 1;
  for (int hh = threadIdx.x; hh < HID; hh += blockDim.x) {
    float s = 0.0f;
    for (int n = 0; n < NN; ++n)
      if (n >= st && n <= en) s += gx[((size_t)b * NN + n) * HID + hh];
    xsum[(size_t)b * HID + hh] = s;
  }
}

__global__ void k_gta(const float* __restrict__ xsum, const float* __restrict__ cm,
                      float* __restrict__ gta) {
  __shared__ float sl[NN];
  __shared__ float sa[NN];
  int b = blockIdx.x;
  int tid = threadIdx.x;
  if (tid < NN) {
    float s = 0.0f;
    const float* cr = cm + ((size_t)b * NN + tid) * HID;
    const float* xr = xsum + (size_t)b * HID;
    for (int hh = 0; hh < HID; ++hh) s += xr[hh] * cr[hh];
    sl[tid] = s;
  }
  __syncthreads();
  if (tid == 0) {
    float mx = -1e30f;
    for (int m = 0; m < NN; ++m) mx = fmaxf(mx, sl[m]);
    float den = 0.0f;
    for (int m = 0; m < NN; ++m) { float e = expf(sl[m] - mx); sa[m] = e; den += e; }
    float inv = 1.0f / den;
    for (int m = 0; m < NN; ++m) sa[m] *= inv;
  }
  __syncthreads();
  for (int hh = tid; hh < HID; hh += blockDim.x) {
    float s = 0.0f;
    for (int m = 0; m < NN; ++m) s += sa[m] * cm[((size_t)b * NN + m) * HID + hh];
    gta[(size_t)b * HID + hh] = s;
  }
}

__global__ void k_fc(const float* __restrict__ af, const float* __restrict__ cf,
                     const float* __restrict__ gta, const float* __restrict__ W,
                     const float* __restrict__ bias, float* __restrict__ out) {
  int t = blockIdx.x * blockDim.x + threadIdx.x;
  if (t >= BB * POL) return;
  int b = t / POL, p = t % POL;
  const float* wr = W + (size_t)p * (3 * HID);
  float s = bias[p];
  for (int i = 0; i < HID; ++i) s += af[(size_t)b * HID + i] * wr[i];
  for (int i = 0; i < HID; ++i) s += cf[(size_t)b * HID + i] * wr[HID + i];
  for (int i = 0; i < HID; ++i) s += gta[(size_t)b * HID + i] * wr[2 * HID + i];
  out[t] = s;
}

// ---------------------------------------------------------------------------
extern "C" void kernel_launch(void* const* d_in, const int* in_sizes, int n_in,
                              void* d_out, int out_size, void* d_ws, size_t ws_size,
                              hipStream_t stream) {
  // params flattened in setup_inputs() insertion order
  const float* embed = (const float*)d_in[0];
  const float* cWih = (const float*)d_in[1];
  const float* cWhh = (const float*)d_in[2];
  const float* cbih = (const float*)d_in[3];
  const float* cbhh = (const float*)d_in[4];
  const float* aWih = (const float*)d_in[5];
  const float* aWhh = (const float*)d_in[6];
  const float* abih = (const float*)d_in[7];
  const float* abhh = (const float*)d_in[8];
  const float* aaWk = (const float*)d_in[9];
  const float* aabk = (const float*)d_in[10];
  const float* aaWq = (const float*)d_in[11];
  const float* aabq = (const float*)d_in[12];
  const float* aaWbl = (const float*)d_in[13];
  const float* aaWp = (const float*)d_in[14];
  const float* aabp = (const float*)d_in[15];
  const float* acWk = (const float*)d_in[16];
  const float* acbk = (const float*)d_in[17];
  const float* acWq = (const float*)d_in[18];
  const float* acbq = (const float*)d_in[19];
  const float* acWbl = (const float*)d_in[20];
  const float* acWp = (const float*)d_in[21];
  const float* acbp = (const float*)d_in[22];
  const float* gWl = (const float*)d_in[23];
  const float* gbl = (const float*)d_in[24];
  const float* gWr = (const float*)d_in[25];
  const float* gbr = (const float*)d_in[26];
  const float* gatt = (const float*)d_in[27];
  const float* gbias = (const float*)d_in[28];
  const float* fcW = (const float*)d_in[29];
  const float* fcb = (const float*)d_in[30];
  const int* text = (const int*)d_in[31];
  const int* aspect = (const int*)d_in[32];
  const int* left = (const int*)d_in[33];
  const float* adj = (const float*)d_in[34];
  const int* esrc = (const int*)d_in[35];
  const int* edst = (const int*)d_in[36];
  float* out = (float*)d_out;
  (void)in_sizes; (void)n_in; (void)out_size; (void)ws_size;

  // byte bump allocator over workspace (256B aligned)
  char* wsp = (char*)d_ws;
  auto allocB = [&](size_t bytes) {
    void* p = (void*)wsp;
    wsp += (bytes + 255) & ~(size_t)255;
    return p;
  };
  auto allocF = [&](size_t nf) { return (float*)allocB(nf * 4); };
  auto allocH = [&](size_t nh) { return (bf16_t*)allocB(nh * 2); };

  // bf16 activations / weights
  bf16_t* ctx_emb = allocH((size_t)NNODE * EMBP);
  bf16_t* asp_emb = allocH((size_t)BB * LAQ * EMBP);
  bf16_t* cWih_p = allocH((size_t)G3 * EMBP);
  bf16_t* aWih_p = allocH((size_t)G3 * EMBP);
  bf16_t* cWhh_b = allocH((size_t)G3 * HID);
  bf16_t* aWhh_b = allocH((size_t)G3 * HID);
  bf16_t* aaWk_b = allocH((size_t)HID * HID);
  bf16_t* aaWq_b = allocH((size_t)HID * HID);
  bf16_t* aaWp_b = allocH((size_t)HID * HID);
  bf16_t* acWk_b = allocH((size_t)HID * HID);
  bf16_t* acWq_b = allocH((size_t)HID * HID);
  bf16_t* acWp_b = allocH((size_t)HID * HID);
  bf16_t* gWl_b = allocH((size_t)GATF * HID);
  bf16_t* gWr_b = allocH((size_t)GATF * HID);
  bf16_t* h_c_b = allocH(BB * HID);
  bf16_t* h_a_b = allocH(BB * HID);
  bf16_t* out_c_b = allocH((size_t)NNODE * HID);
  bf16_t* out_a_b = allocH((size_t)BB * LAQ * HID);
  bf16_t* ctx_pool = allocH(BB * HID);
  bf16_t* asp_pool = allocH(BB * HID);
  bf16_t* ao_b = allocH(BB * HID);
  // fp32 buffers
  float* xg_c = allocF((size_t)NNODE * G3);
  float* xg_a = allocF((size_t)BB * LAQ * G3);
  float* h_c = allocF(BB * HID);
  float* h_a = allocF(BB * HID);
  float* hg  = allocF(BB * G3);
  float* out_c = allocF((size_t)NNODE * HID);
  float* out_a = allocF((size_t)BB * LAQ * HID);
  int* tlen = (int*)allocB(BB * 4);
  int* alen = (int*)allocB(BB * 4);
  int* llen = (int*)allocB(BB * 4);
  float* kx_a = allocF((size_t)BB * LAQ * HID);
  float* kx_c = allocF((size_t)NNODE * HID);
  float* qx = allocF(BB * HID);
  float* qw = allocF(BB * HID);
  float* afin = allocF(BB * HID);
  float* cfin = allocF(BB * HID);
  float* xl = allocF((size_t)NNODE * GATF);
  float* xr = allocF((size_t)NNODE * GATF);
  float* ebuf = allocF((size_t)NE * GH);
  float* gx = allocF((size_t)NNODE * HID);
  float* cm = allocF((size_t)NNODE * HID);
  float* xsum = allocF(BB * HID);
  float* gta = allocF(BB * HID);

  // lengths + embeddings (K-padded bf16) + weight conversion (one-shot)
  k_lengths<<<1, 32, 0, stream>>>(text, aspect, left, tlen, alen, llen);
  k_embed<<<NNODE, 128, 0, stream>>>(embed, text, ctx_emb);
  k_embed<<<BB * LAQ, 128, 0, stream>>>(embed, aspect, asp_emb);
  k_pad_w<<<G3, 128, 0, stream>>>(cWih, cWih_p);
  k_pad_w<<<G3, 128, 0, stream>>>(aWih, aWih_p);
  cvt(cWhh, cWhh_b, G3 * HID, stream);
  cvt(aWhh, aWhh_b, G3 * HID, stream);
  cvt(aaWk, aaWk_b, HID * HID, stream);
  cvt(aaWq, aaWq_b, HID * HID, stream);
  cvt(aaWp, aaWp_b, HID * HID, stream);
  cvt(acWk, acWk_b, HID * HID, stream);
  cvt(acWq, acWq_b, HID * HID, stream);
  cvt(acWp, acWp_b, HID * HID, stream);
  cvt(gWl, gWl_b, GATF * HID, stream);
  cvt(gWr, gWr_b, GATF * HID, stream);

  // GRU input projections (WMMA): xg = x @ Wih^T + bih   (K padded 300->320)
  gemm(ctx_emb, EMBP, cWih_p, EMBP, cbih, xg_c, G3, NNODE, G3, EMBP, stream);
  gemm(asp_emb, EMBP, aWih_p, EMBP, abih, xg_a, G3, BB * LAQ, G3, EMBP, stream);

  hipMemsetAsync(h_c, 0, BB * HID * 4, stream);
  hipMemsetAsync(h_a, 0, BB * HID * 4, stream);
  hipMemsetAsync(h_c_b, 0, BB * HID * 2, stream);
  hipMemsetAsync(h_a_b, 0, BB * HID * 2, stream);

  // sequential GRU recurrences: per-step hg = h @ Whh^T + bhh (WMMA), then gates
  for (int t = 0; t < NN; ++t) {
    gemm(h_c_b, HID, cWhh_b, HID, cbhh, hg, G3, BB, G3, HID, stream);
    k_gru_step<<<BB, 256, 0, stream>>>(xg_c, hg, h_c, h_c_b, out_c, out_c_b, tlen, t, NN);
  }
  for (int t = 0; t < LAQ; ++t) {
    gemm(h_a_b, HID, aWhh_b, HID, abhh, hg, G3, BB, G3, HID, stream);
    k_gru_step<<<BB, 256, 0, stream>>>(xg_a, hg, h_a, h_a_b, out_a, out_a_b, alen, t, LAQ);
  }

  // pools (bf16, feed attention query GEMMs)
  k_pool_mean<<<BB, 256, 0, stream>>>(out_c, tlen, ctx_pool, NN);
  k_pool_mean<<<BB, 256, 0, stream>>>(out_a, alen, asp_pool, LAQ);

  // attention a: keys = aspect GRU outs, query = context_pool
  gemm(out_a_b, HID, aaWk_b, HID, aabk, kx_a, HID, BB * LAQ, HID, HID, stream);
  gemm(ctx_pool, HID, aaWq_b, HID, aabq, qx, HID, BB, HID, HID, stream);
  k_qw<<<(BB * HID + 255) / 256, 256, 0, stream>>>(qx, aaWbl, qw);
  k_att_score<<<BB * AH, 128, 0, stream>>>(qw, kx_a, ao_b, LAQ);
  gemm(ao_b, HID, aaWp_b, HID, aabp, afin, HID, BB, HID, HID, stream);

  // attention c: keys = context GRU outs, query = aspect_pool
  gemm(out_c_b, HID, acWk_b, HID, acbk, kx_c, HID, NNODE, HID, HID, stream);
  gemm(asp_pool, HID, acWq_b, HID, acbq, qx, HID, BB, HID, HID, stream);
  k_qw<<<(BB * HID + 255) / 256, 256, 0, stream>>>(qx, acWbl, qw);
  k_att_score<<<BB * AH, 128, 0, stream>>>(qw, kx_c, ao_b, NN);
  gemm(ao_b, HID, acWp_b, HID, acbp, cfin, HID, BB, HID, HID, stream);

  // GATv2 (dominant GEMMs, WMMA): xl/xr = x @ Wl/Wr^T + b
  gemm(out_c_b, HID, gWl_b, HID, gbl, xl, GATF, NNODE, GATF, HID, stream);
  gemm(out_c_b, HID, gWr_b, HID, gbr, xr, GATF, NNODE, GATF, HID, stream);
  k_edge_e<<<NE, 256, 0, stream>>>(xl, xr, esrc, edst, gatt, ebuf);
  k_gat_agg<<<NNODE, 256, 0, stream>>>(xl, ebuf, esrc, gbias, gx);

  // fusion
  k_cmask<<<NNODE, 256, 0, stream>>>(out_c, adj, llen, alen, cm);
  k_xsum<<<BB, 256, 0, stream>>>(gx, llen, alen, xsum);
  k_gta<<<BB, 512, 0, stream>>>(xsum, cm, gta);
  k_fc<<<1, 128, 0, stream>>>(afin, cfin, gta, fcW, fcb, out);
}
